// AdvancedTemporalGNN_59889023976179
// MI455X (gfx1250) — matrix operand order repaired and checked by reading, not compiled, not run on hardware
//
#include <hip/hip_runtime.h>
#include <math.h>

// ---------------------------------------------------------------------------
// AdvancedTemporalGNN for MI455X (gfx1250).
//  - dense GEMMs: 64x128 block tile, 4x v_wmma_f32_16x16x32_f16 per wave per
//    K-step (A fragment reused), padded LDS to avoid bank conflicts,
//    clamped (branch-free) M-edge handling, global_prefetch_b8 lookahead
//  - attention staging: global_load_async_to_lds_b128 (+ s_wait_asynccnt)
// ---------------------------------------------------------------------------

typedef __attribute__((ext_vector_type(16))) _Float16 v16h;
typedef __attribute__((ext_vector_type(8)))  float    v8f;
typedef __attribute__((address_space(3)))    float    lds_f;

#define NN   20000
#define EE   320000
#define HHH  8
#define HD16 16
#define FINP 64
#define FEA  16
#define BBB  256
#define TTT  50
#define MEMM 512

#define APITCH 40    // halves per As row (80B, 16B-aligned, conflict-free)
#define BPITCH 136   // halves per Bs row (272B, 16B-aligned, conflict-free)

// ---- CDNA5 async copy: 16B global -> LDS, tracked with ASYNCcnt ------------
__device__ __forceinline__ void async_copy16(void* lds_dst, const float* g_src)
{
    lds_f* l = (lds_f*)lds_dst;     // addrspacecast generic -> LDS (32-bit)
    asm volatile("global_load_async_to_lds_b128 %0, %1, off"
                 :: "v"(l), "v"(g_src) : "memory");
}
__device__ __forceinline__ void async_wait0()
{
    asm volatile("s_wait_asynccnt 0x0" ::: "memory");
}

// ---------------------------------------------------------------------------
// WMMA GEMM:  C[M,Nc] = act( A[M,K] @ W[K,Nc] + bias )   (f32 io, f16 mma)
// Requires Kdim % 32 == 0 and Nc % 128 == 0 (true for every call site).
// block 256 threads = 8 waves; macro tile 64(M) x 128(N); K step 32.
// wave w -> M tile (w>>1)*16, N half (w&1)*64 (4 N-subtiles, 4 wmma/step).
// act: 0=none, 1=gelu(tanh), 2=relu
// ---------------------------------------------------------------------------
__global__ __launch_bounds__(256)
void gemm_kernel(const float* __restrict__ A, const float* __restrict__ W,
                 const float* __restrict__ bias, float* __restrict__ C,
                 int Mrows, int Kdim, int Nc, int act)
{
    __shared__ __align__(16) _Float16 As[64 * APITCH];
    __shared__ __align__(16) _Float16 Bs[32 * BPITCH];
    const int tid  = threadIdx.x;
    const int lane = tid & 31;
    const int wave = tid >> 5;
    const int tileM = (wave >> 1) * 16;     // 0..48
    const int tileN = (wave & 1) * 64;      // 0 or 64
    const int rowBase = blockIdx.y * 64;
    const int colBase = blockIdx.x * 128;

    // staging coordinates (fixed per thread)
    const int ar_r = tid >> 2;              // A: row 0..63
    const int ar_k = (tid & 3) << 3;        // A: k offset 0/8/16/24
    const int gr   = rowBase + ar_r;
    const int grc  = (gr < Mrows) ? gr : (Mrows - 1);   // clamp: branch-free
    const int br_k = tid >> 3;              // B: k row 0..31
    const int br_c = (tid & 7) << 4;        // B: col offset 0..112

    const float* aptr = A + (long)grc * Kdim + ar_k;
    const float* bptr = W + (long)br_k * Nc + colBase + br_c;
    const long   bstep = (long)32 * Nc;

    v8f acc[4] = {};
    const int ksteps = Kdim >> 5;
    for (int kt = 0; kt < ksteps; ++kt) {
        // ---- stage A tile 64x32: two b128 loads -> one b128 LDS store ------
        {
            const float4* ap = (const float4*)aptr;
            float4 a0 = ap[0], a1 = ap[1];
            if (kt + 1 < ksteps) __builtin_prefetch(aptr + 32, 0, 1);
            union { _Float16 h[8]; uint4 u; } pk;
            pk.h[0] = (_Float16)a0.x; pk.h[1] = (_Float16)a0.y;
            pk.h[2] = (_Float16)a0.z; pk.h[3] = (_Float16)a0.w;
            pk.h[4] = (_Float16)a1.x; pk.h[5] = (_Float16)a1.y;
            pk.h[6] = (_Float16)a1.z; pk.h[7] = (_Float16)a1.w;
            *(uint4*)(&As[ar_r * APITCH + ar_k]) = pk.u;
        }
        // ---- stage B tile 32x128: four b128 loads -> two b128 LDS stores ---
        {
            const float4* bp = (const float4*)bptr;
            float4 w0 = bp[0], w1 = bp[1], w2 = bp[2], w3 = bp[3];
            if (kt + 1 < ksteps) __builtin_prefetch(bptr + bstep, 0, 1);
            union { _Float16 h[16]; uint4 u[2]; } pb;
            pb.h[0]  = (_Float16)w0.x; pb.h[1]  = (_Float16)w0.y;
            pb.h[2]  = (_Float16)w0.z; pb.h[3]  = (_Float16)w0.w;
            pb.h[4]  = (_Float16)w1.x; pb.h[5]  = (_Float16)w1.y;
            pb.h[6]  = (_Float16)w1.z; pb.h[7]  = (_Float16)w1.w;
            pb.h[8]  = (_Float16)w2.x; pb.h[9]  = (_Float16)w2.y;
            pb.h[10] = (_Float16)w2.z; pb.h[11] = (_Float16)w2.w;
            pb.h[12] = (_Float16)w3.x; pb.h[13] = (_Float16)w3.y;
            pb.h[14] = (_Float16)w3.z; pb.h[15] = (_Float16)w3.w;
            *(uint4*)(&Bs[br_k * BPITCH + br_c])     = pb.u[0];
            *(uint4*)(&Bs[br_k * BPITCH + br_c + 8]) = pb.u[1];
        }
        aptr += 32;
        bptr += bstep;
        __syncthreads();

        // ---- fragments per ISA 7.12.2 (wave32) -----------------------------
        v16h af;
        const int arow = tileM + (lane & 15);
        const int hi   = (lane >> 4) << 3;        // 0 or 8
#pragma unroll
        for (int e = 0; e < 16; ++e) {
            const int kk = (e & 7) + ((e >> 3) << 4) + hi;
            af[e] = As[arow * APITCH + kk];
        }
#pragma unroll
        for (int j = 0; j < 4; ++j) {
            v16h bf;
#pragma unroll
            for (int e = 0; e < 16; ++e) {
                bf[e] = Bs[((e & 7) + ((e >> 3) << 4) + hi) * BPITCH + tileN + j * 16 + (lane & 15)];  // e=K (table order), lane=N
            }
            acc[j] = __builtin_amdgcn_wmma_f32_16x16x32_f16(
                         false, af, false, bf, (short)0, acc[j], false, false);
        }
        __syncthreads();
    }

    // ---- epilogue: rows r / r+8 across lane halves, col = lane&15 ----------
    const int rOff = (lane >> 4) << 3;
#pragma unroll
    for (int j = 0; j < 4; ++j) {
        const int col = colBase + tileN + j * 16 + (lane & 15);
#pragma unroll
        for (int r = 0; r < 8; ++r) {
            const int row = rowBase + tileM + r + rOff;
            if (row < Mrows) {
                float v = acc[j][r] + bias[col];
                if (act == 1) {
                    float x3 = v * v * v;
                    v = 0.5f * v * (1.f + tanhf(0.7978845608f * (v + 0.044715f * x3)));
                } else if (act == 2) {
                    v = fmaxf(v, 0.f);
                }
                C[(long)row * Nc + col] = v;
            }
        }
    }
}

// ---------------------------------------------------------------------------
// LayerNorm over D=128:  Y = act( LN(X (+res)) * g + b ).  one wave per row.
// ---------------------------------------------------------------------------
__global__ __launch_bounds__(256)
void ln_kernel(const float* __restrict__ X, const float* __restrict__ res,
               const float* __restrict__ g, const float* __restrict__ b,
               float* __restrict__ Y, int rows, int relu)
{
    const int wave = threadIdx.x >> 5, lane = threadIdx.x & 31;
    const int row = blockIdx.x * 8 + wave;
    if (row >= rows) return;
    const float* xr = X + (long)row * 128;
    const float* rr = res ? res + (long)row * 128 : nullptr;
    float v[4], s = 0.f, s2 = 0.f;
#pragma unroll
    for (int i = 0; i < 4; ++i) {
        const int c = lane * 4 + i;
        float t = xr[c] + (rr ? rr[c] : 0.f);
        v[i] = t; s += t; s2 += t * t;
    }
#pragma unroll
    for (int m = 16; m >= 1; m >>= 1) {
        s  += __shfl_xor(s,  m, 32);
        s2 += __shfl_xor(s2, m, 32);
    }
    const float mean = s * (1.f / 128.f);
    const float var  = s2 * (1.f / 128.f) - mean * mean;
    const float rstd = rsqrtf(var + 1e-5f);
#pragma unroll
    for (int i = 0; i < 4; ++i) {
        const int c = lane * 4 + i;
        float o = (v[i] - mean) * rstd * g[c] + b[c];
        if (relu) o = fmaxf(o, 0.f);
        Y[(long)row * 128 + c] = o;
    }
}

// ---------------------------------------------------------------------------
// Edge attention (3 passes), segment softmax over dst with atomics.
// ---------------------------------------------------------------------------
__device__ inline void atomicMaxF(float* addr, float v)
{
    int old = __float_as_int(*addr);
    while (__int_as_float(old) < v) {
        int assumed = old;
        old = atomicCAS((int*)addr, assumed, __float_as_int(v));
        if (old == assumed) break;
    }
}

__global__ void init_stats_kernel(float* NM, float* NS, int n)
{
    int i = blockIdx.x * 256 + threadIdx.x;
    if (i < n) { NM[i] = -1e30f; NS[i] = 0.f; }
}

__device__ __forceinline__ void load16(const float* p, float* out)  // 64B aligned
{
    const float4* p4 = (const float4*)p;
#pragma unroll
    for (int j = 0; j < 4; ++j) {
        float4 v = p4[j];
        out[j * 4 + 0] = v.x; out[j * 4 + 1] = v.y;
        out[j * 4 + 2] = v.z; out[j * 4 + 3] = v.w;
    }
}

__global__ __launch_bounds__(256)
void edge_logits_kernel(const float* __restrict__ Q, const float* __restrict__ Kn,
                        const float* __restrict__ ea, const float* __restrict__ we,
                        const float* __restrict__ be,
                        const int* __restrict__ src, const int* __restrict__ dst,
                        float* __restrict__ EL, float* __restrict__ NM)
{
    __shared__ float sWe[FEA * 128];
    __shared__ float sBe[128];
    for (int i = threadIdx.x; i < FEA * 128; i += 256) sWe[i] = we[i];
    if (threadIdx.x < 128) sBe[threadIdx.x] = be[threadIdx.x];
    __syncthreads();
    const long idx = (long)blockIdx.x * 256 + threadIdx.x;
    if (idx >= (long)EE * HHH) return;
    const int e = (int)(idx >> 3), h = (int)(idx & 7);
    const int s = src[e], d = dst[e];
    float eav[16], qv[16], kv[16];
    load16(ea + (long)e * FEA, eav);
    load16(Q  + (long)d * 128 + h * 16, qv);
    load16(Kn + (long)s * 128 + h * 16, kv);
    float acc = 0.f;
#pragma unroll
    for (int dd = 0; dd < HD16; ++dd) {
        const int col = h * 16 + dd;
        float ep = sBe[col];
#pragma unroll
        for (int j = 0; j < FEA; ++j) ep += eav[j] * sWe[j * 128 + col];
        acc += qv[dd] * (kv[dd] + ep);
    }
    const float logit = acc * 0.25f;   // 1/sqrt(16)
    EL[idx] = logit;
    atomicMaxF(&NM[(long)d * 8 + h], logit);
}

__global__ void edge_exp_kernel(const int* __restrict__ dst, float* __restrict__ EL,
                                const float* __restrict__ NM, float* __restrict__ NS)
{
    const long idx = (long)blockIdx.x * 256 + threadIdx.x;
    if (idx >= (long)EE * HHH) return;
    const int e = (int)(idx >> 3), h = (int)(idx & 7);
    const int d = dst[e];
    const float w = expf(EL[idx] - NM[(long)d * 8 + h]);
    EL[idx] = w;
    atomicAdd(&NS[(long)d * 8 + h], w);
}

__global__ __launch_bounds__(256)
void edge_agg_kernel(const float* __restrict__ V, const float* __restrict__ ea,
                     const float* __restrict__ we, const float* __restrict__ be,
                     const int* __restrict__ src, const int* __restrict__ dst,
                     const float* __restrict__ EL, const float* __restrict__ NS,
                     float* __restrict__ Out)
{
    __shared__ float sWe[FEA * 128];
    __shared__ float sBe[128];
    for (int i = threadIdx.x; i < FEA * 128; i += 256) sWe[i] = we[i];
    if (threadIdx.x < 128) sBe[threadIdx.x] = be[threadIdx.x];
    __syncthreads();
    const long idx = (long)blockIdx.x * 256 + threadIdx.x;
    if (idx >= (long)EE * HHH) return;
    const int e = (int)(idx >> 3), h = (int)(idx & 7);
    const int s = src[e], d = dst[e];
    const float coef = EL[idx] / (NS[(long)d * 8 + h] + 1e-16f);
    float eav[16], vv[16];
    load16(ea + (long)e * FEA, eav);
    load16(V  + (long)s * 128 + h * 16, vv);
#pragma unroll
    for (int dd = 0; dd < HD16; ++dd) {
        const int col = h * 16 + dd;
        float ep = sBe[col];
#pragma unroll
        for (int j = 0; j < FEA; ++j) ep += eav[j] * sWe[j * 128 + col];
        atomicAdd(&Out[(long)d * 128 + col], coef * (vv[dd] + ep));
    }
}

// ---------------------------------------------------------------------------
// seq = h[node_seq] + time_encoding
// ---------------------------------------------------------------------------
__global__ void gather_te_kernel(const float* __restrict__ h, const int* __restrict__ nseq,
                                 const float* __restrict__ ts, const float* __restrict__ fs,
                                 const float* __restrict__ ph, float* __restrict__ out)
{
    const int idx = blockIdx.x * 256 + threadIdx.x;
    if (idx >= BBB * TTT * 128) return;
    const int bt = idx >> 7, c = idx & 127;
    const int node = nseq[bt];
    const float t = ts[bt];
    const int i = c & 63;
    const float div = expf((float)i * -0.14391156f) * fs[i];  // -ln(10000)/64
    const float ang = t * div + ph[i];
    const float te = (c < 64) ? sinf(ang) : cosf(ang);
    out[idx] = h[(long)node * 128 + c] + te;
}

// ---------------------------------------------------------------------------
// MHA attention core (per b,h block; T=50, HD=16); async staging to LDS
// ---------------------------------------------------------------------------
__global__ __launch_bounds__(64)
void mha_attn_kernel(const float* __restrict__ Q, const float* __restrict__ Kb,
                     const float* __restrict__ Vb, float* __restrict__ Ctx)
{
    const int b = blockIdx.x >> 3, h = blockIdx.x & 7;
    __shared__ __align__(16) float qs[TTT * 16];
    __shared__ __align__(16) float ks[TTT * 16];
    __shared__ __align__(16) float vs[TTT * 16];
    for (int i = threadIdx.x; i < TTT * 4; i += 64) {     // 200 x 16B per array
        const int t = i >> 2, p4 = (i & 3) << 2;
        const long o = ((long)(b * TTT + t)) * 128 + h * 16 + p4;
        async_copy16(&qs[i * 4], Q  + o);
        async_copy16(&ks[i * 4], Kb + o);
        async_copy16(&vs[i * 4], Vb + o);
    }
    async_wait0();
    __syncthreads();
    const int t = threadIdx.x;
    if (t >= TTT) return;
    float sc[TTT];
    float mx = -1e30f;
    for (int j = 0; j < TTT; ++j) {
        float s = 0.f;
#pragma unroll
        for (int d = 0; d < 16; ++d) s += qs[t * 16 + d] * ks[j * 16 + d];
        s *= 0.25f;
        sc[j] = s; mx = fmaxf(mx, s);
    }
    float sum = 0.f;
    for (int j = 0; j < TTT; ++j) { sc[j] = expf(sc[j] - mx); sum += sc[j]; }
    const float inv = 1.f / sum;
    for (int d = 0; d < 16; ++d) {
        float acc = 0.f;
        for (int j = 0; j < TTT; ++j) acc += sc[j] * vs[j * 16 + d];
        Ctx[((long)(b * TTT + t)) * 128 + h * 16 + d] = acc * inv;
    }
}

// ---------------------------------------------------------------------------
// elementwise ops
// ---------------------------------------------------------------------------
__global__ void add_kernel(const float* a, const float* b, float* y, int n)
{
    int i = blockIdx.x * 256 + threadIdx.x;
    if (i < n) y[i] = a[i] + b[i];
}
__global__ void add_relu_kernel(const float* a, const float* b, float* y, int n)
{
    int i = blockIdx.x * 256 + threadIdx.x;
    if (i < n) y[i] = fmaxf(a[i] + b[i], 0.f);
}

// ---------------------------------------------------------------------------
// causal dilated conv via im2col:  COL[(b,t), k*128+c] = X[b, t-(2-k)*dil, c]
// ---------------------------------------------------------------------------
__global__ void im2col_kernel(const float* __restrict__ X, float* __restrict__ COL, int dil)
{
    const int idx = blockIdx.x * 256 + threadIdx.x;
    if (idx >= BBB * TTT * 384) return;
    const int row = idx / 384, j = idx % 384;
    const int k = j >> 7, c = j & 127;
    const int b = row / TTT, t = row % TTT;
    const int tt = t - (2 - k) * dil;
    COL[idx] = (tt >= 0) ? X[((long)(b * TTT + tt)) * 128 + c] : 0.f;
}

// Wcol[(k*128+i)*128 + o] = w[o, i, k]   (w flat: o*384 + i*3 + k)
__global__ void repack_w_kernel(const float* __restrict__ w, float* __restrict__ Wcol)
{
    const int idx = blockIdx.x * 256 + threadIdx.x;
    if (idx >= 384 * 128) return;
    const int o = idx & 127;
    const int ki = idx >> 7;
    const int k = ki >> 7, i = ki & 127;
    Wcol[idx] = w[(long)o * 384 + i * 3 + k];
}

__global__ void pool_kernel(const float* __restrict__ X, float* __restrict__ P)
{
    const int idx = blockIdx.x * 256 + threadIdx.x;
    if (idx >= BBB * 128) return;
    const int b = idx >> 7, c = idx & 127;
    float s = 0.f;
    for (int t = 0; t < TTT; ++t) s += X[((long)(b * TTT + t)) * 128 + c];
    P[idx] = s * (1.f / TTT);
}

// ---------------------------------------------------------------------------
// memory read attention: per-b block, 4 heads (one wave each), 512 slots
// ---------------------------------------------------------------------------
__global__ __launch_bounds__(128)
void mem_attn_kernel(const float* __restrict__ Qm, const float* __restrict__ Km,
                     const float* __restrict__ Vm, float* __restrict__ R)
{
    const int b = blockIdx.x;
    __shared__ __align__(16) float qs[128];
    __shared__ float sc[4 * MEMM];
    const int tid = threadIdx.x;
    if (tid < 32) async_copy16(&qs[tid * 4], Qm + (long)b * 128 + tid * 4);
    async_wait0();
    __syncthreads();
    const int h = tid >> 5, lane = tid & 31;
    float lmax = -1e30f;
    for (int m = lane; m < MEMM; m += 32) {
        float s = 0.f;
#pragma unroll
        for (int d = 0; d < 32; ++d) s += qs[h * 32 + d] * Km[(long)m * 128 + h * 32 + d];
        s *= 0.17677669f;   // 1/sqrt(32)
        sc[h * MEMM + m] = s;
        lmax = fmaxf(lmax, s);
    }
#pragma unroll
    for (int m = 16; m >= 1; m >>= 1) lmax = fmaxf(lmax, __shfl_xor(lmax, m, 32));
    float lsum = 0.f;
    for (int m = lane; m < MEMM; m += 32) {
        const float w = expf(sc[h * MEMM + m] - lmax);
        sc[h * MEMM + m] = w;
        lsum += w;
    }
#pragma unroll
    for (int m = 16; m >= 1; m >>= 1) lsum += __shfl_xor(lsum, m, 32);
    const float inv = 1.f / lsum;
    __syncthreads();
    const int d = lane;
    float acc = 0.f;
    for (int m = 0; m < MEMM; ++m) acc += sc[h * MEMM + m] * Vm[(long)m * 128 + h * 32 + d];
    R[(long)b * 128 + h * 32 + d] = acc * inv;
}

__global__ void concat_kernel(const float* a, const float* b, float* y)
{
    const int idx = blockIdx.x * 256 + threadIdx.x;
    if (idx >= BBB * 256) return;
    const int r = idx >> 8, c = idx & 255;
    y[idx] = (c < 128) ? a[(long)r * 128 + c] : b[(long)r * 128 + (c - 128)];
}

// ---------------------------------------------------------------------------
// host-side launch
// ---------------------------------------------------------------------------
static inline void gemm(hipStream_t st, const float* A, const float* W,
                        const float* bias, float* C, int M, int K, int Nc, int act)
{
    dim3 g(Nc / 128, (M + 63) / 64);
    gemm_kernel<<<g, 256, 0, st>>>(A, W, bias, C, M, K, Nc, act);
}
static inline void ln(hipStream_t st, const float* X, const float* res,
                      const float* g, const float* b, float* Y, int rows, int relu)
{
    ln_kernel<<<(rows + 7) / 8, 256, 0, st>>>(X, res, g, b, Y, rows, relu);
}

extern "C" void kernel_launch(void* const* d_in, const int* in_sizes, int n_in,
                              void* d_out, int out_size, void* d_ws, size_t ws_size,
                              hipStream_t stream)
{
    // -------- input map: jax pytree flatten (dict keys sorted) ---------------
    // 0 edge_attr  1 edge_index  2 node_seq
    // 3..10 attn: bk bo bq bv wk wo wq wv
    // 11 enc_b  12 enc_w
    // 13+g*18: be bk bq bs bv fb1 fb2 g1 g2 gb1 gb2 w1 w2 we wk wq ws wv
    // 67..75 mem: bk bo bq bv mem wk wo wq wv
    // 76 out_b  77 out_w  78 te_fs  79 te_ph
    // 80+t*8: b1 b2 c1b c1w c2b c2w g1 g2
    // 104 timestamps  105 x
    auto F = [&](int i) { return (const float*)d_in[i]; };
    const int* ei   = (const int*)d_in[1];
    const int* nseq = (const int*)d_in[2];
    const int* src = ei;
    const int* dst = ei + EE;

    // -------- workspace arena (floats) --------------------------------------
    float* ws = (float*)d_ws;
    float* W0  = ws;                                  // N*512 (multi-purpose)
    float* Bf0 = W0  + (size_t)NN * 512;              // N*128 : h
    float* Bf1 = Bf0 + (size_t)NN * 128;
    float* Bf2 = Bf1 + (size_t)NN * 128;
    float* Bf3 = Bf2 + (size_t)NN * 128;
    float* EL  = Bf3 + (size_t)NN * 128;              // E*8
    float* NM  = EL  + (size_t)EE * 8;                // N*8
    float* NS  = NM  + (size_t)NN * 8;                // N*8
    float* SM  = NS  + (size_t)NN * 8;                // small region
    float* WCOL = SM;                 // 384*128
    float* POOL = WCOL + 384 * 128;   // 256*128
    float* RMEM = POOL + BBB * 128;   // 256*128
    float* RET  = RMEM + BBB * 128;   // 256*128
    float* CAT  = RET  + BBB * 128;   // 256*256
    float* KM   = CAT  + BBB * 256;   // 512*128
    float* VM   = KM   + MEMM * 128;  // 512*128
    float* QM   = VM   + MEMM * 128;  // 256*128

    const int SEQN = BBB * TTT * 128;           // 1,638,400
    const int EH   = EE * HHH;                  // 2,560,000

    // ======================= 1. node encoder ================================
    gemm(stream, F(105), F(12), F(11), Bf0, NN, FINP, 128, 0);

    // ======================= 2. graph transformer layers ====================
    float* Wq = W0;
    float* Wk = W0 + (size_t)NN * 128;
    float* Wv = W0 + (size_t)2 * NN * 128;
    for (int g = 0; g < 3; ++g) {
        const int p = 13 + g * 18;
        gemm(stream, Bf0, F(p + 15), F(p + 2), Wq,  NN, 128, 128, 0);  // wq,bq
        gemm(stream, Bf0, F(p + 14), F(p + 1), Wk,  NN, 128, 128, 0);  // wk,bk
        gemm(stream, Bf0, F(p + 17), F(p + 4), Wv,  NN, 128, 128, 0);  // wv,bv
        gemm(stream, Bf0, F(p + 16), F(p + 3), Bf2, NN, 128, 128, 0);  // skip ws,bs

        init_stats_kernel<<<(NN * 8 + 255) / 256, 256, 0, stream>>>(NM, NS, NN * 8);
        edge_logits_kernel<<<EH / 256, 256, 0, stream>>>(Wq, Wk, F(0), F(p + 13), F(p + 0),
                                                         src, dst, EL, NM);
        edge_exp_kernel<<<EH / 256, 256, 0, stream>>>(dst, EL, NM, NS);
        edge_agg_kernel<<<EH / 256, 256, 0, stream>>>(Wv, F(0), F(p + 13), F(p + 0),
                                                      src, dst, EL, NS, Bf2);

        ln(stream, Bf2, Bf0, F(p + 7), F(p + 9), Bf1, NN, 0);          // h1 = ln(conv+x)
        gemm(stream, Bf1, F(p + 11), F(p + 5), W0,  NN, 128, 512, 1);  // gelu(h1@w1+fb1)
        gemm(stream, W0,  F(p + 12), F(p + 6), Bf3, NN, 512, 128, 0);  // @w2+fb2
        ln(stream, Bf1, Bf3, F(p + 8), F(p + 10), Bf0, NN, 0);         // h = ln(h1+f)
    }

    // ======================= 3. sequence gather + time enc ==================
    float* SEQ = W0;                    // graph phase of W0 is over
    float* TA  = W0 + (size_t)SEQN;
    float* TB  = W0 + (size_t)2 * SEQN;
    float* COL = W0 + (size_t)3 * SEQN; // BBB*TTT*384
    gather_te_kernel<<<SEQN / 256, 256, 0, stream>>>(Bf0, nseq, F(104), F(78), F(79), SEQ);

    // ======================= 4. MHA =========================================
    float* MQ = Bf0; float* MK = Bf1; float* MV = Bf2; float* MC = Bf3; float* MO = EL;
    gemm(stream, SEQ, F(9),  F(5), MQ, BBB * TTT, 128, 128, 0);
    gemm(stream, SEQ, F(7),  F(3), MK, BBB * TTT, 128, 128, 0);
    gemm(stream, SEQ, F(10), F(6), MV, BBB * TTT, 128, 128, 0);
    mha_attn_kernel<<<BBB * HHH, 64, 0, stream>>>(MQ, MK, MV, MC);
    gemm(stream, MC, F(8), F(4), MO, BBB * TTT, 128, 128, 0);
    add_kernel<<<SEQN / 256, 256, 0, stream>>>(SEQ, MO, SEQ, SEQN);

    // ======================= 5. temporal conv blocks ========================
    for (int t = 0; t < 3; ++t) {
        const int p = 80 + t * 8;
        const int dil = 1 << t;
        repack_w_kernel<<<(384 * 128 + 255) / 256, 256, 0, stream>>>(F(p + 3), WCOL);
        im2col_kernel<<<(BBB * TTT * 384 + 255) / 256, 256, 0, stream>>>(SEQ, COL, dil);
        gemm(stream, COL, WCOL, F(p + 2), TA, BBB * TTT, 384, 128, 0);
        ln(stream, TA, nullptr, F(p + 6), F(p + 0), TA, BBB * TTT, 1);
        repack_w_kernel<<<(384 * 128 + 255) / 256, 256, 0, stream>>>(F(p + 5), WCOL);
        im2col_kernel<<<(BBB * TTT * 384 + 255) / 256, 256, 0, stream>>>(TA, COL, dil);
        gemm(stream, COL, WCOL, F(p + 4), TB, BBB * TTT, 384, 128, 0);
        ln(stream, TB, nullptr, F(p + 7), F(p + 1), TB, BBB * TTT, 1);
        add_relu_kernel<<<SEQN / 256, 256, 0, stream>>>(TB, SEQ, SEQ, SEQN);
    }

    // ======================= 6. pool + memory read ==========================
    pool_kernel<<<(BBB * 128 + 255) / 256, 256, 0, stream>>>(SEQ, POOL);
    gemm(stream, F(71), F(72), F(67), KM, MEMM, 128, 128, 0);   // Km = mem@wk+bk
    gemm(stream, F(71), F(75), F(70), VM, MEMM, 128, 128, 0);   // Vm = mem@wv+bv
    gemm(stream, POOL,  F(74), F(69), QM, BBB, 128, 128, 0);    // Q  = pooled@wq+bq
    mem_attn_kernel<<<BBB, 128, 0, stream>>>(QM, KM, VM, RMEM);
    gemm(stream, RMEM, F(73), F(68), RET, BBB, 128, 128, 0);    // ret = r@wo+bo

    // ======================= 7. output head =================================
    concat_kernel<<<(BBB * 256 + 255) / 256, 256, 0, stream>>>(POOL, RET, CAT);
    gemm(stream, CAT, F(77), F(76), (float*)d_out, BBB, 256, 128, 0);
}
